// SimpleRNN_5729486372970
// MI455X (gfx1250) — matrix operand order repaired and checked by reading, not compiled, not run on hardware
//
#include <hip/hip_runtime.h>
#include <hip/hip_bf16.h>

#define B_ 128
#define S_ 1024
#define I_ 512
#define H_ 1024
#define O_ 512

typedef __attribute__((ext_vector_type(16))) __bf16 v16bf;
typedef __attribute__((ext_vector_type(8)))  __bf16 v8bf;
typedef __attribute__((ext_vector_type(4)))  __bf16 v4bf;
typedef __attribute__((ext_vector_type(8)))  float  v8f;

// Build the 16x32 bf16 A-fragment for one lane.
// Layout: lanes 0-15 hold K = kk+0..15; lanes 16-31 hold
// K = kk+16..31. Caller passes p = row_base + kk + half*16.
static __device__ __forceinline__ v16bf make_a_frag(const __bf16* __restrict__ p) {
    v8bf lo = *(const v8bf*)p;          // 16B aligned
    v8bf hi = *(const v8bf*)(p + 8);    // +16B
    return __builtin_shufflevector(lo, hi, 0,1,2,3,4,5,6,7,8,9,10,11,12,13,14,15);
}

// ---------------------------------------------------------------- convert
__global__ void f32_to_bf16_v4(const float* __restrict__ src,
                               __bf16* __restrict__ dst, long n4) {
    long i      = (long)blockIdx.x * blockDim.x + threadIdx.x;
    long stride = (long)gridDim.x * blockDim.x;
    for (; i < n4; i += stride) {
        float4 f = ((const float4*)src)[i];
        v4bf o = { (__bf16)f.x, (__bf16)f.y, (__bf16)f.z, (__bf16)f.w };
        ((v4bf*)dst)[i] = o;
    }
}

// ---------------------------------------------------------------- init
__global__ void init_state(__bf16* __restrict__ h0, __bf16* __restrict__ h1,
                           int* __restrict__ cnt) {
    int i = blockIdx.x * blockDim.x + threadIdx.x;
    if (i == 0) *cnt = 0;
    const int n = B_ * H_;
    for (int j = i; j < n; j += gridDim.x * blockDim.x) {
        h0[j] = (__bf16)0.0f;
        h1[j] = (__bf16)0.0f;
    }
}

// ---------------------------------------------------------------- xp GEMM
// xp[s][b][n] = sum_k x[b][s][k] * W_ih[n][k] + b_ih[n] + b_hh[n]
// M = B*S (row r = b*S+s), N = H, K = I. Per WG: 128x64 tile, W_ih tile in LDS.
// Every 128-row M tile lies inside a single batch index (128 divides S), so
// the b/s split is a per-block constant and the epilogue is a strided store.
__global__ __launch_bounds__(256) void xp_gemm(
    const __bf16* __restrict__ xb,    // [B*S, I]
    const __bf16* __restrict__ Wih,   // [H, I]
    const float*  __restrict__ b_ih,
    const float*  __restrict__ b_hh,
    float* __restrict__ xp) {         // [S, B, H]
    const int mTile = blockIdx.x;           // 0..1023
    const int n0    = blockIdx.y * 64;      // 0..15 -> N base
    const int lane  = threadIdx.x & 31;
    const int wave  = threadIdx.x >> 5;     // 0..7
    const int half  = lane >> 4;
    const int l16   = lane & 15;

    __shared__ __attribute__((aligned(16))) __bf16 ldsW[64 * I_];  // 64 KB
    {
        const v8bf* src = (const v8bf*)(Wih + (size_t)n0 * I_);
        v8bf* dst = (v8bf*)ldsW;
        for (int i = threadIdx.x; i < 64 * I_ / 8; i += 256) dst[i] = src[i];
    }
    __syncthreads();

    const int mBase = mTile * 128 + wave * 16;
    const int arow  = mBase + l16;
    v8f acc[4] = {};

    for (int kk = 0; kk < I_; kk += 32) {
        v16bf a = make_a_frag(xb + (size_t)arow * I_ + kk + half * 16);
        const int kb = kk + half * 16;
        #pragma unroll
        for (int j = 0; j < 4; ++j) {
            v16bf b = *(const v16bf*)(&ldsW[(size_t)(j * 16 + l16) * I_ + kb]);
            acc[j] = __builtin_amdgcn_wmma_f32_16x16x32_bf16(
                false, a, false, b, (short)0, acc[j], false, false);
        }
    }

    // epilogue: hoisted batch/seq split, strided scatter into [S,B,H]
    const int bb    = mBase >> 10;              // mBase / S_  (constant per block)
    const int sBase = (mBase & (S_ - 1)) + half * 8;
    float* __restrict__ outp =
        xp + ((size_t)sBase * B_ + bb) * H_;    // row s=sBase, batch bb
    #pragma unroll
    for (int j = 0; j < 4; ++j) {
        const int n    = n0 + j * 16 + l16;
        const float bv = b_ih[n] + b_hh[n];
        #pragma unroll
        for (int v = 0; v < 8; ++v) {
            outp[(size_t)v * (B_ * H_) + n] = acc[j][v] + bv;  // s = sBase + v
        }
    }
}

// ---------------------------------------------------------------- recurrence
// Persistent kernel, 32 WGs x 256 threads. Each WG owns a 32-column slice of h.
// W_hh slice lives in LDS for the whole sequence. Global counter barrier per step.
// xp slices (512 KB/step) stream from HBM -> prefetch next step's slice early.
__global__ __launch_bounds__(256) void rnn_recurrence(
    const __bf16* __restrict__ Whh,   // [H, H] row-major bf16
    const float*  __restrict__ xp,    // [S, B, H]
    __bf16* __restrict__ h0,          // ping (zero-initialized)
    __bf16* __restrict__ h1,          // pong
    int* __restrict__ bar) {
    const int n0   = blockIdx.x * 32;
    const int lane = threadIdx.x & 31;
    const int wave = threadIdx.x >> 5;
    const int half = lane >> 4;
    const int l16  = lane & 15;

    __shared__ __attribute__((aligned(16))) __bf16 ldsW[32 * H_];  // 64 KB
    {
        const v8bf* src = (const v8bf*)(Whh + (size_t)n0 * H_);
        v8bf* dst = (v8bf*)ldsW;
        for (int i = threadIdx.x; i < 32 * H_ / 8; i += 256) dst[i] = src[i];
    }
    __syncthreads();

    const int mBase = wave * 16;            // 8 waves cover all 128 batch rows
    const int arow  = mBase + l16;

    // per-lane prefetch target inside this wave's [16 x 32] xp tile:
    // lane covers row (mBase + l16), 64B segment (half) of the 128B col span
    const size_t pfOff = (size_t)(mBase + l16) * H_ + n0 + half * 16;

    for (int t = 0; t < S_; ++t) {
        const __bf16* __restrict__ hcur = (t & 1) ? h1 : h0;
        __bf16*       __restrict__ hnxt = (t & 1) ? h0 : h1;

        // hide HBM latency of this step's xp slice behind the K loop
        __builtin_prefetch(xp + (size_t)t * (B_ * H_) + pfOff, 0, 1);
        if (t + 1 < S_)  // and warm next step's slice too
            __builtin_prefetch(xp + (size_t)(t + 1) * (B_ * H_) + pfOff, 0, 1);

        v8f acc0 = {};
        v8f acc1 = {};
        #pragma unroll 4
        for (int kk = 0; kk < H_; kk += 32) {
            v16bf a = make_a_frag(hcur + (size_t)arow * H_ + kk + half * 16);
            const int kb = kk + half * 16;
            v16bf b0 = *(const v16bf*)(&ldsW[(size_t)(l16)      * H_ + kb]);
            v16bf b1 = *(const v16bf*)(&ldsW[(size_t)(16 + l16) * H_ + kb]);
            acc0 = __builtin_amdgcn_wmma_f32_16x16x32_bf16(
                false, a, false, b0, (short)0, acc0, false, false);
            acc1 = __builtin_amdgcn_wmma_f32_16x16x32_bf16(
                false, a, false, b1, (short)0, acc1, false, false);
        }

        const float* __restrict__ xpt = xp + (size_t)t * (B_ * H_);
        #pragma unroll
        for (int v = 0; v < 8; ++v) {
            const int m  = mBase + half * 8 + v;
            const int na = n0 + l16;
            const int nb = n0 + 16 + l16;
            float f0 = tanhf(acc0[v] + xpt[(size_t)m * H_ + na]);
            float f1 = tanhf(acc1[v] + xpt[(size_t)m * H_ + nb]);
            hnxt[(size_t)m * H_ + na] = (__bf16)f0;
            hnxt[(size_t)m * H_ + nb] = (__bf16)f1;
        }

        // device-wide step barrier (monotonic counter, no reset needed)
        __threadfence();
        __syncthreads();
        if (threadIdx.x == 0) {
            __hip_atomic_fetch_add(bar, 1, __ATOMIC_ACQ_REL, __HIP_MEMORY_SCOPE_AGENT);
            const int target = 32 * (t + 1);
            while (__hip_atomic_load(bar, __ATOMIC_ACQUIRE, __HIP_MEMORY_SCOPE_AGENT) < target) {
                __builtin_amdgcn_s_sleep(2);
            }
        }
        __syncthreads();
    }
    // S_ = 1024 (even) -> final hidden state ends up in h0
}

// ---------------------------------------------------------------- fc
// out[b][n] = sum_k hT[b][k] * W_fc[n][k] + b_fc[n];  M=128, N=512, K=1024
__global__ __launch_bounds__(256) void fc_gemm(
    const __bf16* __restrict__ hT,    // [B, H] bf16 (final hidden, in h0)
    const __bf16* __restrict__ Wfc,   // [O, H]
    const float*  __restrict__ b_fc,
    float* __restrict__ out) {        // [B, O]
    const int n0   = blockIdx.x * 64;
    const int lane = threadIdx.x & 31;
    const int wave = threadIdx.x >> 5;
    const int half = lane >> 4;
    const int l16  = lane & 15;

    const int mBase = wave * 16;
    const int arow  = mBase + l16;
    v8f acc[4] = {};

    for (int kk = 0; kk < H_; kk += 32) {
        v16bf a = make_a_frag(hT + (size_t)arow * H_ + kk + half * 16);
        const int kb = kk + half * 16;
        #pragma unroll
        for (int j = 0; j < 4; ++j) {
            const int n = n0 + j * 16 + l16;
            v16bf b = *(const v16bf*)(Wfc + (size_t)n * H_ + kb);
            acc[j] = __builtin_amdgcn_wmma_f32_16x16x32_bf16(
                false, a, false, b, (short)0, acc[j], false, false);
        }
    }

    #pragma unroll
    for (int j = 0; j < 4; ++j) {
        const int n = n0 + j * 16 + l16;
        #pragma unroll
        for (int v = 0; v < 8; ++v) {
            const int m = mBase + half * 8 + v;
            out[(size_t)m * O_ + n] = acc[j][v] + b_fc[n];
        }
    }
}

// ---------------------------------------------------------------- launch
extern "C" void kernel_launch(void* const* d_in, const int* in_sizes, int n_in,
                              void* d_out, int out_size, void* d_ws, size_t ws_size,
                              hipStream_t stream) {
    const float* x    = (const float*)d_in[0];   // [B,S,I]
    const float* Wih  = (const float*)d_in[1];   // [H,I]
    const float* Whh  = (const float*)d_in[2];   // [H,H]
    const float* b_ih = (const float*)d_in[3];   // [H]
    const float* b_hh = (const float*)d_in[4];   // [H]
    const float* Wfc  = (const float*)d_in[5];   // [O,H]
    const float* b_fc = (const float*)d_in[6];   // [O]
    float* out = (float*)d_out;                  // [B,O]

    // workspace carve (256B aligned)
    uint8_t* ws  = (uint8_t*)d_ws;
    size_t   off = 0;
    auto carve = [&](size_t bytes) -> void* {
        void* p = ws + off;
        off = (off + bytes + 255) & ~(size_t)255;
        return p;
    };
    float*  xp     = (float*) carve((size_t)S_ * B_ * H_ * 4);   // 512 MB
    __bf16* xb     = (__bf16*)carve((size_t)B_ * S_ * I_ * 2);   // 128 MB
    __bf16* Wih_b  = (__bf16*)carve((size_t)H_ * I_ * 2);
    __bf16* Whh_b  = (__bf16*)carve((size_t)H_ * H_ * 2);
    __bf16* Wfc_b  = (__bf16*)carve((size_t)O_ * H_ * 2);
    __bf16* h0     = (__bf16*)carve((size_t)B_ * H_ * 2);
    __bf16* h1     = (__bf16*)carve((size_t)B_ * H_ * 2);
    int*    bar    = (int*)   carve(256);
    (void)ws_size; (void)in_sizes; (void)n_in; (void)out_size;

    // 1) f32 -> bf16 conversions
    {
        long n4;
        n4 = (long)B_ * S_ * I_ / 4;
        f32_to_bf16_v4<<<4096, 256, 0, stream>>>(x, xb, n4);
        n4 = (long)H_ * I_ / 4;
        f32_to_bf16_v4<<<512, 256, 0, stream>>>(Wih, Wih_b, n4);
        n4 = (long)H_ * H_ / 4;
        f32_to_bf16_v4<<<1024, 256, 0, stream>>>(Whh, Whh_b, n4);
        n4 = (long)O_ * H_ / 4;
        f32_to_bf16_v4<<<512, 256, 0, stream>>>(Wfc, Wfc_b, n4);
    }

    // 2) zero h ping/pong + barrier counter
    init_state<<<512, 256, 0, stream>>>(h0, h1, bar);

    // 3) input projection GEMM (all timesteps at once)
    xp_gemm<<<dim3(1024, 16), 256, 0, stream>>>(xb, Wih_b, b_ih, b_hh, xp);

    // 4) persistent recurrence (32 WGs, W_hh slice resident in LDS)
    rnn_recurrence<<<32, 256, 0, stream>>>(Whh_b, xp, h0, h1, bar);

    // 5) final FC on last hidden state (ends in h0)
    fc_gemm<<<8, 256, 0, stream>>>(h0, Wfc_b, b_fc, out);
}
